// Attention__38886633898436
// MI455X (gfx1250) — matrix operand order repaired and checked by reading, not compiled, not run on hardware
//
#include <hip/hip_runtime.h>
#include <hip/hip_bf16.h>
#include <math.h>

// ---------------------------------------------------------------------------
// Attention block on gfx1250 (CDNA5): qkv GEMM -> flash attention -> proj
// - all matmuls: v_wmma_f32_16x16x32_bf16 (f32 accumulate)
// - f32->bf16 via v_cvt_pk_bf16_f32 when available
// - K tile staging via Tensor Data Mover (tensor_load_to_lds) when available
// ---------------------------------------------------------------------------

#define N_TOK   4096
#define DIM     768
#define QKV_DIM 2304   // 3 * DIM
#define NHEAD   12
#define HD      64     // head dim
#define SCALE   0.125f // HD^-0.5

typedef __bf16 bf16_t;
typedef bf16_t        v16bf __attribute__((ext_vector_type(16)));
typedef bf16_t        v2bf  __attribute__((ext_vector_type(2)));
typedef float         v8f   __attribute__((ext_vector_type(8)));
typedef unsigned int  v4u   __attribute__((ext_vector_type(4)));
typedef int           v8i   __attribute__((ext_vector_type(8)));
typedef int           v4i   __attribute__((ext_vector_type(4)));

union Frag16 {               // 16 bf16 elements = 8 dwords (one WMMA A/B operand)
    v16bf        v;
    unsigned int u[8];
};

#if defined(__gfx1250__) && __has_builtin(__builtin_amdgcn_tensor_load_to_lds)
#define HAVE_TDM 1
#else
#define HAVE_TDM 0
#endif

__device__ __forceinline__ v8f v8f_zero() {
    v8f z;
#pragma unroll
    for (int i = 0; i < 8; ++i) z[i] = 0.0f;
    return z;
}

__device__ __forceinline__ unsigned short f32_to_bf16(float f) {
    union { float f; unsigned int u; } c; c.f = f;
    unsigned int u = c.u;
    u += 0x7FFFu + ((u >> 16) & 1u);   // round-to-nearest-even
    return (unsigned short)(u >> 16);
}

// packed f32x2 -> bf16x2 in one dword (v_cvt_pk_bf16_f32 on CDNA5)
__device__ __forceinline__ unsigned int pack_bf16(float lo, float hi) {
#if __has_builtin(__builtin_amdgcn_cvt_pk_bf16_f32)
    union { v2bf v; unsigned int u; } c;
    c.v = __builtin_amdgcn_cvt_pk_bf16_f32(lo, hi);
    return c.u;
#else
    return (unsigned int)f32_to_bf16(lo) | ((unsigned int)f32_to_bf16(hi) << 16);
#endif
}

__device__ __forceinline__ float half_reduce_max(float v) {
#pragma unroll
    for (int m = 1; m < 16; m <<= 1) v = fmaxf(v, __shfl_xor(v, m, 32));
    return v;
}
__device__ __forceinline__ float half_reduce_sum(float v) {
#pragma unroll
    for (int m = 1; m < 16; m <<= 1) v += __shfl_xor(v, m, 32);
    return v;
}

// ISA 7.12.2: 16-bit A-matrix 16x32.  lane: M = lane%16, half = lane/16.
__device__ __forceinline__ int a_frag_k(int g, int half) {
    return (g < 4) ? (2 * g + 8 * half) : (16 + 2 * (g - 4) + 8 * half);
}
// 16-bit B-matrix 32x16. lane: N = lane%16, half = lane/16.
__device__ __forceinline__ int b_frag_k(int g, int half) {
    return (g < 4) ? (2 * g + 8 * half) : (16 + 2 * (g - 4) + 8 * half);
}

// ---------------------------------------------------------------------------
// Kernel 1: qkv = x @ w_qkv^T  (4096x768 @ 768x2304), output bf16, Q pre-scaled
// 256 threads (8 waves), tile 128x128, K-step 32; wave = 32x64 sub-tile.
// ---------------------------------------------------------------------------
__global__ __launch_bounds__(256)
void qkv_gemm_kernel(const float* __restrict__ x,
                     const float* __restrict__ w,          // [2304][768]
                     unsigned short* __restrict__ qkv_bf)  // [4096][2304] bf16
{
    __shared__ unsigned short aS[128][40];  // 128 rows x 32 K (pad->40)
    __shared__ unsigned short bS[128][40];

    const int tid  = threadIdx.x;
    const int lane = tid & 31;
    const int wave = tid >> 5;
    const int wm   = wave >> 1;
    const int wn   = wave & 1;
    const int half = lane >> 4;
    const int l16  = lane & 15;
    const int m0   = blockIdx.y * 128;
    const int n0   = blockIdx.x * 128;

    v8f acc[2][4];
#pragma unroll
    for (int i = 0; i < 2; ++i)
#pragma unroll
        for (int j = 0; j < 4; ++j) acc[i][j] = v8f_zero();

    for (int kt = 0; kt < DIM; kt += 32) {
#pragma unroll
        for (int p = 0; p < 4; ++p) {
            int chunk = p * 256 + tid;          // 1024 chunks of 4 floats
            int row   = chunk >> 3;
            int c4    = (chunk & 7) * 4;
            float4 av = *(const float4*)(x + (size_t)(m0 + row) * DIM + kt + c4);
            float4 bv = *(const float4*)(w + (size_t)(n0 + row) * DIM + kt + c4);
            *(uint2*)&aS[row][c4] = make_uint2(pack_bf16(av.x, av.y), pack_bf16(av.z, av.w));
            *(uint2*)&bS[row][c4] = make_uint2(pack_bf16(bv.x, bv.y), pack_bf16(bv.z, bv.w));
        }
        if (kt + 32 < DIM) {   // global_prefetch_b8 hint for next tile
            __builtin_prefetch(x + (size_t)(m0 + (tid >> 1)) * DIM + kt + 32, 0, 1);
        }
        __syncthreads();

        Frag16 af[2], bfr[4];
#pragma unroll
        for (int i = 0; i < 2; ++i) {
            int ar = wm * 32 + i * 16 + l16;
#pragma unroll
            for (int g = 0; g < 8; ++g)
                af[i].u[g] = *(const unsigned int*)&aS[ar][a_frag_k(g, half)];
        }
#pragma unroll
        for (int j = 0; j < 4; ++j) {
            int bc = wn * 64 + j * 16 + l16;
#pragma unroll
            for (int g = 0; g < 8; ++g)
                bfr[j].u[g] = *(const unsigned int*)&bS[bc][b_frag_k(g, half)];
        }
#pragma unroll
        for (int i = 0; i < 2; ++i)
#pragma unroll
            for (int j = 0; j < 4; ++j)
                acc[i][j] = __builtin_amdgcn_wmma_f32_16x16x32_bf16(
                    false, af[i].v, false, bfr[j].v, (short)0, acc[i][j], false, false);
        __syncthreads();
    }

#pragma unroll
    for (int i = 0; i < 2; ++i)
#pragma unroll
        for (int j = 0; j < 4; ++j) {
            int col = n0 + wn * 64 + j * 16 + l16;
            float s = (col < DIM) ? SCALE : 1.0f;   // Q columns pre-scaled hd^-0.5
#pragma unroll
            for (int g = 0; g < 8; ++g) {
                int row = m0 + wm * 32 + i * 16 + g + 8 * half;
                qkv_bf[(size_t)row * QKV_DIM + col] = f32_to_bf16(acc[i][j][g] * s);
            }
        }
}

// ---------------------------------------------------------------------------
// Kernel 2: flash attention. blockIdx = (query block of 64, head).
// 128 threads = 4 waves; each wave owns 16 query rows.  Key blocks of 64.
// K tile staged by the Tensor Data Mover when available.
// ---------------------------------------------------------------------------
__global__ __launch_bounds__(128)
void attn_kernel(const unsigned short* __restrict__ qkv,      // [4096][2304] bf16
                 unsigned short* __restrict__ attn_out)       // [4096][768]  bf16
{
    __shared__ unsigned short kS [64][72];      // K block  [key][hd]   (TDM-padded)
    __shared__ unsigned short vtS[64][72];      // V block transposed [hd][key]
    __shared__ unsigned short pS [4][16][72];   // per-wave P staging [row][key]

    const int tid  = threadIdx.x;
    const int lane = tid & 31;
    const int wave = tid >> 5;
    const int half = lane >> 4;
    const int l16  = lane & 15;
    const int h    = blockIdx.y;
    const int q0   = blockIdx.x * 64;

    Frag16 qf[2];
    const unsigned short* qrow = qkv + (size_t)(q0 + wave * 16 + l16) * QKV_DIM + h * HD;
#pragma unroll
    for (int f = 0; f < 2; ++f)
#pragma unroll
        for (int g = 0; g < 8; ++g)
            qf[f].u[g] = *(const unsigned int*)(qrow + 32 * f + a_frag_k(g, half));

    float mrow[8], lrow[8];
#pragma unroll
    for (int g = 0; g < 8; ++g) { mrow[g] = -3.0e38f; lrow[g] = 0.0f; }
    v8f o[4];
#pragma unroll
    for (int j = 0; j < 4; ++j) o[j] = v8f_zero();

    for (int kb = 0; kb < N_TOK / 64; ++kb) {
        // ---- stage K tile -------------------------------------------------
#if HAVE_TDM
        if (wave == 0) {
            // D# (CDNA5 ISA): 2D tile, data_size=4B,
            // tile 32x64 dwords, row stride 1152 dwords, LDS pad 4dw / 32dw.
            unsigned long long ga = (unsigned long long)(uintptr_t)
                (qkv + (size_t)(kb * 64) * QKV_DIM + DIM + h * HD);
            unsigned int lds_off = (unsigned int)(uintptr_t)&kS[0][0];
            v4u g0;
            g0[0] = 1u;                                        // count=1 (valid)
            g0[1] = lds_off;                                   // lds_addr
            g0[2] = (unsigned int)(ga & 0xFFFFFFFFu);          // global_addr[31:0]
            g0[3] = (unsigned int)((ga >> 32) & 0x01FFFFFFu)   // global_addr[56:32]
                    | 0x80000000u;                             // type=2 ("image")
            v8i g1;
            g1[0] = (int)((2u << 16)      // data_size: 4 bytes
                        | (1u << 20)      // pad_enable
                        | (4u << 22)      // pad_interval: every 32 dwords
                        | (3u << 25));    // pad_amount:  4 dwords
            g1[1] = (int)(32u << 16);     // tensor_dim0 = 32 (dwords/row)
            g1[2] = (int)(64u << 16);     // tensor_dim1 = 64 (rows)
            g1[3] = (int)(32u << 16);     // tile_dim0   = 32
            g1[4] = 64;                   // tile_dim1   = 64 (tile_dim2 = 0)
            g1[5] = 1152;                 // tensor_dim0_stride (QKV_DIM/2 dwords)
            g1[6] = 0;
            g1[7] = 0;
            v4i gz;
#pragma unroll
            for (int i = 0; i < 4; ++i) gz[i] = 0;
#if __clang_major__ >= 23
            v8i gz8;
#pragma unroll
            for (int i = 0; i < 8; ++i) gz8[i] = 0;
            __builtin_amdgcn_tensor_load_to_lds(g0, g1, gz, gz, gz8, 0);
#else
            __builtin_amdgcn_tensor_load_to_lds(g0, g1, gz, gz, 0);
#endif
#if __has_builtin(__builtin_amdgcn_s_wait_tensorcnt)
            __builtin_amdgcn_s_wait_tensorcnt(0);
#else
            asm volatile("s_wait_tensorcnt 0x0" ::: "memory");
#endif
        }
        // ---- stage V^T (manual: transpose while staging) ------------------
#pragma unroll
        for (int p = 0; p < 8; ++p) {
            int chunk = p * 128 + tid;
            int row   = chunk >> 4;
            int c4    = (chunk & 15) * 4;
            uint2 vv = *(const uint2*)(qkv + (size_t)(kb * 64 + row) * QKV_DIM
                                       + 2 * DIM + h * HD + c4);
            vtS[c4 + 0][row] = (unsigned short)(vv.x & 0xFFFFu);
            vtS[c4 + 1][row] = (unsigned short)(vv.x >> 16);
            vtS[c4 + 2][row] = (unsigned short)(vv.y & 0xFFFFu);
            vtS[c4 + 3][row] = (unsigned short)(vv.y >> 16);
        }
#else
        // ---- fallback: manual K + V^T staging -----------------------------
#pragma unroll
        for (int p = 0; p < 8; ++p) {
            int chunk = p * 128 + tid;
            int row   = chunk >> 4;
            int c4    = (chunk & 15) * 4;
            size_t base = (size_t)(kb * 64 + row) * QKV_DIM + h * HD + c4;
            uint2 kv = *(const uint2*)(qkv + base + DIM);
            uint2 vv = *(const uint2*)(qkv + base + 2 * DIM);
            *(uint2*)&kS[row][c4] = kv;
            vtS[c4 + 0][row] = (unsigned short)(vv.x & 0xFFFFu);
            vtS[c4 + 1][row] = (unsigned short)(vv.x >> 16);
            vtS[c4 + 2][row] = (unsigned short)(vv.y & 0xFFFFu);
            vtS[c4 + 3][row] = (unsigned short)(vv.y >> 16);
        }
#endif
        __syncthreads();

        // ---- S = Q K^T  (4 tiles of 16 keys, hd=64 -> 2 WMMA each) --------
        v8f s[4];
#pragma unroll
        for (int t = 0; t < 4; ++t) {
            s[t] = v8f_zero();
#pragma unroll
            for (int f = 0; f < 2; ++f) {
                Frag16 kf;
#pragma unroll
                for (int g = 0; g < 8; ++g)
                    kf.u[g] = *(const unsigned int*)&kS[t * 16 + l16][32 * f + b_frag_k(g, half)];
                s[t] = __builtin_amdgcn_wmma_f32_16x16x32_bf16(
                    false, qf[f].v, false, kf.v, (short)0, s[t], false, false);
            }
        }

        // ---- online softmax (row = g + 8*half across a 16-lane half) ------
        float alpha[8], mnew[8];
#pragma unroll
        for (int g = 0; g < 8; ++g) {
            float v = fmaxf(fmaxf(s[0][g], s[1][g]), fmaxf(s[2][g], s[3][g]));
            float mb = half_reduce_max(v);
            mnew[g]  = fmaxf(mrow[g], mb);
            alpha[g] = __expf(mrow[g] - mnew[g]);
            mrow[g]  = mnew[g];
        }
#pragma unroll
        for (int t = 0; t < 4; ++t)
#pragma unroll
            for (int g = 0; g < 8; ++g)
                s[t][g] = __expf(s[t][g] - mnew[g]);
#pragma unroll
        for (int g = 0; g < 8; ++g) {
            float rs = half_reduce_sum(s[0][g] + s[1][g] + s[2][g] + s[3][g]);
            lrow[g] = lrow[g] * alpha[g] + rs;
        }
#pragma unroll
        for (int j = 0; j < 4; ++j)
#pragma unroll
            for (int g = 0; g < 8; ++g)
                o[j][g] *= alpha[g];

        // ---- P -> LDS (C-layout -> A-layout reshape) ----------------------
#pragma unroll
        for (int t = 0; t < 4; ++t)
#pragma unroll
            for (int g = 0; g < 8; ++g)
                pS[wave][g + 8 * half][t * 16 + l16] = f32_to_bf16(s[t][g]);
        // same-wave LDS ops are in-order: stores land before the frag loads.

        // ---- O += P V -----------------------------------------------------
#pragma unroll
        for (int f = 0; f < 2; ++f) {
            Frag16 pf;
#pragma unroll
            for (int g = 0; g < 8; ++g)
                pf.u[g] = *(const unsigned int*)&pS[wave][l16][32 * f + a_frag_k(g, half)];
#pragma unroll
            for (int j = 0; j < 4; ++j) {
                Frag16 vf;
#pragma unroll
                for (int g = 0; g < 8; ++g)
                    vf.u[g] = *(const unsigned int*)&vtS[j * 16 + l16][32 * f + b_frag_k(g, half)];
                o[j] = __builtin_amdgcn_wmma_f32_16x16x32_bf16(
                    false, pf.v, false, vf.v, (short)0, o[j], false, false);
            }
        }
        __syncthreads();   // protect kS/vtS before next restage
    }

#pragma unroll
    for (int j = 0; j < 4; ++j)
#pragma unroll
        for (int g = 0; g < 8; ++g) {
            int row = q0 + wave * 16 + g + 8 * half;
            int col = h * HD + j * 16 + l16;
            attn_out[(size_t)row * DIM + col] = f32_to_bf16(o[j][g] / lrow[g]);
        }
}

// ---------------------------------------------------------------------------
// Kernel 3: out = attn @ w_proj^T + b_proj  (f32 output)
// ---------------------------------------------------------------------------
__global__ __launch_bounds__(256)
void proj_gemm_kernel(const unsigned short* __restrict__ a_bf,  // [4096][768] bf16
                      const float* __restrict__ w,              // [768][768]
                      const float* __restrict__ bias,           // [768]
                      float* __restrict__ out)                  // [4096][768]
{
    __shared__ unsigned short aS[128][40];
    __shared__ unsigned short bS[128][40];

    const int tid  = threadIdx.x;
    const int lane = tid & 31;
    const int wave = tid >> 5;
    const int wm   = wave >> 1;
    const int wn   = wave & 1;
    const int half = lane >> 4;
    const int l16  = lane & 15;
    const int m0   = blockIdx.y * 128;
    const int n0   = blockIdx.x * 128;

    v8f acc[2][4];
#pragma unroll
    for (int i = 0; i < 2; ++i)
#pragma unroll
        for (int j = 0; j < 4; ++j) acc[i][j] = v8f_zero();

    for (int kt = 0; kt < DIM; kt += 32) {
#pragma unroll
        for (int p = 0; p < 4; ++p) {
            int chunk = p * 256 + tid;
            int row   = chunk >> 3;
            int c4    = (chunk & 7) * 4;
            *(uint2*)&aS[row][c4] =
                *(const uint2*)(a_bf + (size_t)(m0 + row) * DIM + kt + c4);
            float4 bv = *(const float4*)(w + (size_t)(n0 + row) * DIM + kt + c4);
            *(uint2*)&bS[row][c4] = make_uint2(pack_bf16(bv.x, bv.y), pack_bf16(bv.z, bv.w));
        }
        if (kt + 32 < DIM) {
            __builtin_prefetch(w + (size_t)(n0 + (tid >> 1)) * DIM + kt + 32, 0, 1);
        }
        __syncthreads();

        Frag16 af[2], bfr[4];
#pragma unroll
        for (int i = 0; i < 2; ++i) {
            int ar = wm * 32 + i * 16 + l16;
#pragma unroll
            for (int g = 0; g < 8; ++g)
                af[i].u[g] = *(const unsigned int*)&aS[ar][a_frag_k(g, half)];
        }
#pragma unroll
        for (int j = 0; j < 4; ++j) {
            int bc = wn * 64 + j * 16 + l16;
#pragma unroll
            for (int g = 0; g < 8; ++g)
                bfr[j].u[g] = *(const unsigned int*)&bS[bc][b_frag_k(g, half)];
        }
#pragma unroll
        for (int i = 0; i < 2; ++i)
#pragma unroll
            for (int j = 0; j < 4; ++j)
                acc[i][j] = __builtin_amdgcn_wmma_f32_16x16x32_bf16(
                    false, af[i].v, false, bfr[j].v, (short)0, acc[i][j], false, false);
        __syncthreads();
    }

#pragma unroll
    for (int i = 0; i < 2; ++i)
#pragma unroll
        for (int j = 0; j < 4; ++j) {
            int col = n0 + wn * 64 + j * 16 + l16;
            float b = bias[col];
#pragma unroll
            for (int g = 0; g < 8; ++g) {
                int row = m0 + wm * 32 + i * 16 + g + 8 * half;
                out[(size_t)row * DIM + col] = acc[i][j][g] + b;
            }
        }
}

// ---------------------------------------------------------------------------
extern "C" void kernel_launch(void* const* d_in, const int* in_sizes, int n_in,
                              void* d_out, int out_size, void* d_ws, size_t ws_size,
                              hipStream_t stream) {
    (void)in_sizes; (void)n_in; (void)out_size; (void)ws_size;
    const float* x      = (const float*)d_in[0];
    const float* w_qkv  = (const float*)d_in[1];
    const float* w_proj = (const float*)d_in[2];
    const float* b_proj = (const float*)d_in[3];
    float* out = (float*)d_out;

    unsigned short* qkv_bf  = (unsigned short*)d_ws;             // 4096*2304 bf16
    unsigned short* attn_bf = qkv_bf + (size_t)N_TOK * QKV_DIM;  // 4096*768  bf16

    qkv_gemm_kernel<<<dim3(QKV_DIM / 128, N_TOK / 128), 256, 0, stream>>>(
        x, w_qkv, qkv_bf);
    attn_kernel<<<dim3(N_TOK / 64, NHEAD), 128, 0, stream>>>(
        qkv_bf, attn_bf);
    proj_gemm_kernel<<<dim3(DIM / 128, N_TOK / 128), 256, 0, stream>>>(
        attn_bf, w_proj, b_proj, out);
}
